// PIR_NCDE_32641751449783
// MI455X (gfx1250) — matrix operand order repaired and checked by reading, not compiled, not run on hardware
//
#include <hip/hip_runtime.h>
#include <hip/hip_bf16.h>
#include <cstdint>
#include <cstddef>

// ---------------------------------------------------------------------------
// Types for CDNA5 WMMA (gfx1250, wave32)
// ---------------------------------------------------------------------------
typedef __bf16 bf16;
typedef __attribute__((ext_vector_type(16))) __bf16 v16bf;
typedef __attribute__((ext_vector_type(8)))  float  v8f;

union FragU { uint4 q[2]; v16bf v; };

__device__ inline v8f wmma_bf(v16bf a, v16bf b, v8f c) {
  // D = A(16x32) * B(32x16) + C(16x16), bf16 inputs, fp32 accumulate
  return __builtin_amdgcn_wmma_f32_16x16x32_bf16(false, a, false, b, (short)0, c,
                                                 false, false);
}

// Async global->LDS copy of 16 bytes per lane (GLOBAL_LOAD_ASYNC_TO_LDS_B128,
// tracked by ASYNCcnt). Low 32 bits of a generic shared pointer are the LDS
// byte address (ISA 10.2).
__device__ inline void async_g2l_b128(const void* gsrc, void* ldst) {
  uint64_t ga = (uint64_t)(uintptr_t)gsrc;
  uint32_t la = (uint32_t)(uintptr_t)ldst;
  asm volatile("global_load_async_to_lds_b128 %0, %1, off"
               :: "v"(la), "v"(ga) : "memory");
}
__device__ inline void async_wait0() {
  asm volatile("s_wait_asynccnt 0" ::: "memory");
}

// A-matrix 16x32 bf16 fragment (this code's k order, same as B): lane = row (lane&15),
// VGPR0..7 hold K = 16*half + {0..15} (contiguous).
__device__ inline v16bf load_a_frag(const bf16* A, int lda, int row0, int kb, int lane) {
  int r = row0 + (lane & 15);
  int half = lane >> 4;
  const bf16* p = A + (size_t)r * lda + kb + half * 16;
  FragU f;
  f.q[0] = *(const uint4*)(p);
  f.q[1] = *(const uint4*)(p + 8);
  return f.v;
}

// B-matrix 32x16 bf16 fragment: lane = column (lane&15),
// VGPR0..7 hold K = 16*half + {0..15} (contiguous), from row-major W (N x K).
__device__ inline v16bf load_b_frag(const bf16* W, int ldw, int n0, int kb, int lane) {
  int n = n0 + (lane & 15);
  int half = lane >> 4;
  const bf16* p = W + (size_t)n * ldw + kb + half * 16;
  FragU f;
  f.q[0] = *(const uint4*)(p);
  f.q[1] = *(const uint4*)(p + 8);
  return f.v;
}

__device__ inline float act_apply(float x, int act) {
  if (act == 1) return fmaxf(x, 0.f);
  if (act == 2) return tanhf(x);
  if (act == 3) return 0.5f * x * (1.f + erff(x * 0.70710678118654752f)); // exact gelu
  return x;
}

// ---------------------------------------------------------------------------
// Generic GEMM: out = act(A(MxK,bf16) @ W(NxK,bf16)^T + bias)
// grid (N/64, M/32), block 128 (4 waves). The 32xK A slab is staged once into
// LDS with async global->LDS b128 copies (slab is contiguous in global).
// ---------------------------------------------------------------------------
__global__ __launch_bounds__(128) void k_gemm(
    const bf16* __restrict__ A, const bf16* __restrict__ W,
    const float* __restrict__ bias, float* __restrict__ outF,
    bf16* __restrict__ outB, int M, int N, int K, int act)
{
  __shared__ __align__(16) bf16 As[32 * 256];       // K <= 256
  int lane = threadIdx.x & 31, wave = threadIdx.x >> 5;
  int mR = blockIdx.y * 32;
  int n0 = blockIdx.x * 64 + wave * 16;

  {
    const bf16* src0 = A + (size_t)mR * K;          // contiguous 32*K slab
    int vecs = (32 * K) >> 3;                       // 16B chunks
    for (int i = threadIdx.x; i < vecs; i += 128)
      async_g2l_b128(src0 + (size_t)i * 8, As + i * 8);
    async_wait0();
  }
  __syncthreads();

  v8f acc0 = {}, acc1 = {};
  for (int kb = 0; kb < K; kb += 32) {
    if (kb + 32 < K)
      __builtin_prefetch(W + (size_t)(n0 + (lane & 15)) * K + kb + 32, 0, 1);
    v16bf bfr = load_b_frag(W, K, n0, kb, lane);
    v16bf a0  = load_a_frag(As, K, 0,  kb, lane);
    v16bf a1  = load_a_frag(As, K, 16, kb, lane);
    acc0 = wmma_bf(a0, bfr, acc0);
    acc1 = wmma_bf(a1, bfr, acc1);
  }
  int cl = lane & 15, half = lane >> 4;
  int col = n0 + cl;
  float bv = bias ? bias[col] : 0.f;
#pragma unroll
  for (int v = 0; v < 8; ++v) {
    int r0 = mR + v + 8 * half;
    int r1 = r0 + 16;
    float x0 = act_apply(acc0[v] + bv, act);
    float x1 = act_apply(acc1[v] + bv, act);
    if (outF) { outF[(size_t)r0 * N + col] = x0;        outF[(size_t)r1 * N + col] = x1; }
    if (outB) { outB[(size_t)r0 * N + col] = (bf16)x0;  outB[(size_t)r1 * N + col] = (bf16)x1; }
  }
}

// ---------------------------------------------------------------------------
// Fused 3-layer MLP for the RK4 scan: zz = z + alpha*kprev;
// h0=relu(zz f0); h1=relu(h0 m0); h2=relu(h1 m1) -> global bf16 (32x256).
// One workgroup, 256 threads (8 waves).
// ---------------------------------------------------------------------------
__device__ inline void mlp_stage(const bf16* A, int K, const bf16* __restrict__ W,
                                 const float* __restrict__ bias, int N,
                                 bf16* outLds, bf16* __restrict__ outG,
                                 int lane, int wave)
{
  for (int nt = wave; nt < (N >> 4); nt += 8) {
    int n0 = nt * 16;
    v8f acc0 = {}, acc1 = {};
    for (int kb = 0; kb < K; kb += 32) {
      v16bf bfr = load_b_frag(W, K, n0, kb, lane);
      v16bf a0  = load_a_frag(A, K, 0,  kb, lane);
      v16bf a1  = load_a_frag(A, K, 16, kb, lane);
      acc0 = wmma_bf(a0, bfr, acc0);
      acc1 = wmma_bf(a1, bfr, acc1);
    }
    int cl = lane & 15, half = lane >> 4;
    int col = n0 + cl;
    float bv = bias[col];
#pragma unroll
    for (int v = 0; v < 8; ++v) {
      int r0 = v + 8 * half, r1 = r0 + 16;
      float x0 = fmaxf(acc0[v] + bv, 0.f);
      float x1 = fmaxf(acc1[v] + bv, 0.f);
      if (outLds) { outLds[r0 * N + col] = (bf16)x0; outLds[r1 * N + col] = (bf16)x1; }
      else        { outG[r0 * N + col]   = (bf16)x0; outG[r1 * N + col]   = (bf16)x1; }
    }
  }
}

__global__ __launch_bounds__(256) void k_mlp3(
    const float* __restrict__ z, const float* __restrict__ kprev, float alpha,
    const bf16* __restrict__ f0W, const float* __restrict__ f0b,
    const bf16* __restrict__ m0W, const float* __restrict__ m0b,
    const bf16* __restrict__ m1W, const float* __restrict__ m1b,
    bf16* __restrict__ h2out)
{
  __shared__ __align__(16) bf16 zz[32 * 128];
  __shared__ __align__(16) bf16 hA[32 * 256];
  __shared__ __align__(16) bf16 hB[32 * 256];
  int tid = threadIdx.x;
  for (int i = tid; i < 32 * 128; i += 256) {
    float v = z[i];
    if (kprev) v += alpha * kprev[i];
    zz[i] = (bf16)v;
  }
  __syncthreads();
  int lane = tid & 31, wave = tid >> 5;
  mlp_stage(zz, 128, f0W, f0b, 256, hA, nullptr, lane, wave);
  __syncthreads();
  mlp_stage(hA, 256, m0W, m0b, 256, hB, nullptr, lane, wave);
  __syncthreads();
  mlp_stage(hB, 256, m1W, m1b, 256, nullptr, h2out, lane, wave);
}

// ---------------------------------------------------------------------------
// Fused f4 GEMM + tanh + einsum (scan variant): fz viewed (H,C) as h*64+c
// k[b,h] = sum_c tanh(f4(h2)[b, h*64+c]) * dxdt[b,c]
// dxdt built from coeffs[:, t] and frac. grid 32 blocks x 128 thr; wave <-> h.
// h2 (32x256) staged to LDS via async copies, overlapped with dx setup.
// ---------------------------------------------------------------------------
__global__ __launch_bounds__(128) void k_f4c_scan(
    const bf16* __restrict__ h2, const bf16* __restrict__ W,
    const float* __restrict__ bias, const float* __restrict__ coeffs,
    int t, float frac, float* __restrict__ kout)
{
  __shared__ float dx[32 * 64];
  __shared__ __align__(16) bf16 Ah[32 * 256];
  int tid = threadIdx.x;
  for (int i = tid; i < 1024; i += 128)             // 32x256 bf16 = 1024 x 16B
    async_g2l_b128(h2 + (size_t)i * 8, Ah + i * 8);
  for (int i = tid; i < 2048; i += 128) {
    int b = i >> 6, c = i & 63;
    const float* cf = coeffs + ((size_t)(b * 255 + t)) * 256;
    float cb = cf[64 + c], cc = cf[128 + c], cd = cf[192 + c];
    dx[i] = cb + 2.f * cc * frac + 3.f * cd * (frac * frac);
  }
  async_wait0();
  __syncthreads();
  int lane = tid & 31, wave = tid >> 5;
  int h = blockIdx.x * 4 + wave;
  v8f acc[2][4] = {};
  for (int kb = 0; kb < 256; kb += 32) {
    v16bf a0 = load_a_frag(Ah, 256, 0,  kb, lane);
    v16bf a1 = load_a_frag(Ah, 256, 16, kb, lane);
#pragma unroll
    for (int j = 0; j < 4; ++j) {
      v16bf bfr = load_b_frag(W, 256, h * 64 + j * 16, kb, lane);
      acc[0][j] = wmma_bf(a0, bfr, acc[0][j]);
      acc[1][j] = wmma_bf(a1, bfr, acc[1][j]);
    }
  }
  int cl = lane & 15, half = lane >> 4;
  float ra[2][8] = {};
#pragma unroll
  for (int j = 0; j < 4; ++j) {
    int c = j * 16 + cl;
    float bv = bias[h * 64 + c];
#pragma unroll
    for (int v = 0; v < 8; ++v) {
      int r0 = v + 8 * half;
      ra[0][v] += tanhf(acc[0][j][v] + bv) * dx[r0 * 64 + c];
      ra[1][v] += tanhf(acc[1][j][v] + bv) * dx[(r0 + 16) * 64 + c];
    }
  }
#pragma unroll
  for (int m = 0; m < 2; ++m)
#pragma unroll
    for (int v = 0; v < 8; ++v) {
      float s = ra[m][v];
      s += __shfl_xor(s, 1, 32); s += __shfl_xor(s, 2, 32);
      s += __shfl_xor(s, 4, 32); s += __shfl_xor(s, 8, 32);
      ra[m][v] = s;
    }
  if (cl == 0) {
#pragma unroll
    for (int m = 0; m < 2; ++m)
#pragma unroll
      for (int v = 0; v < 8; ++v) {
        int r = m * 16 + v + 8 * half;
        kout[r * 128 + h] = ra[m][v];
      }
  }
}

// ---------------------------------------------------------------------------
// Fused f4 GEMM + tanh + einsum (sd variant): flat vector viewed (C,H):
// sd[bt,h] = sum_c tanh(f4(h2)[bt, c*128 + h]) * dxdt[bt,c]
// grid (8192/32, 128/4), block 128; wave <-> h. A slab async-staged to LDS.
// ---------------------------------------------------------------------------
__global__ __launch_bounds__(128) void k_f4c_sd(
    const bf16* __restrict__ h2, const bf16* __restrict__ W,
    const float* __restrict__ bias, const float* __restrict__ dxdt,
    float* __restrict__ sd)
{
  __shared__ float dx[32 * 64];
  __shared__ __align__(16) bf16 Ah[32 * 256];
  int tid = threadIdx.x;
  int mR = blockIdx.x * 32;
  {
    const bf16* src0 = h2 + (size_t)mR * 256;
    for (int i = tid; i < 1024; i += 128)
      async_g2l_b128(src0 + (size_t)i * 8, Ah + i * 8);
  }
  for (int i = tid; i < 2048; i += 128)
    dx[i] = dxdt[(size_t)(mR + (i >> 6)) * 64 + (i & 63)];
  async_wait0();
  __syncthreads();
  int lane = tid & 31, wave = tid >> 5;
  int h = blockIdx.y * 4 + wave;
  v8f acc[2][4] = {};
  int cl = lane & 15, half = lane >> 4;
  for (int kb = 0; kb < 256; kb += 32) {
    v16bf a0 = load_a_frag(Ah, 256, 0,  kb, lane);
    v16bf a1 = load_a_frag(Ah, 256, 16, kb, lane);
#pragma unroll
    for (int j = 0; j < 4; ++j) {
      int r = (j * 16 + cl) * 128 + h;               // W row = c*H + h
      const bf16* p = W + (size_t)r * 256 + kb + half * 16;
      FragU f; f.q[0] = *(const uint4*)(p); f.q[1] = *(const uint4*)(p + 8);
      acc[0][j] = wmma_bf(a0, f.v, acc[0][j]);
      acc[1][j] = wmma_bf(a1, f.v, acc[1][j]);
    }
  }
  float ra[2][8] = {};
#pragma unroll
  for (int j = 0; j < 4; ++j) {
    int c = j * 16 + cl;
    float bv = bias[(size_t)c * 128 + h];
#pragma unroll
    for (int v = 0; v < 8; ++v) {
      int r0 = v + 8 * half;
      ra[0][v] += tanhf(acc[0][j][v] + bv) * dx[r0 * 64 + c];
      ra[1][v] += tanhf(acc[1][j][v] + bv) * dx[(r0 + 16) * 64 + c];
    }
  }
#pragma unroll
  for (int m = 0; m < 2; ++m)
#pragma unroll
    for (int v = 0; v < 8; ++v) {
      float s = ra[m][v];
      s += __shfl_xor(s, 1, 32); s += __shfl_xor(s, 2, 32);
      s += __shfl_xor(s, 4, 32); s += __shfl_xor(s, 8, 32);
      ra[m][v] = s;
    }
  if (cl == 0) {
#pragma unroll
    for (int m = 0; m < 2; ++m)
#pragma unroll
      for (int v = 0; v < 8; ++v) {
        int r = mR + m * 16 + v + 8 * half;
        sd[(size_t)r * 128 + h] = ra[m][v];
      }
  }
}

// ---------------------------------------------------------------------------
// Attention (S=32 per sequence). One block of 32 lanes per (n, head).
// qkv layout (S, N, 3E); output bf16 (S, N, E) with o[s,n,head*hd+d].
// ---------------------------------------------------------------------------
__global__ __launch_bounds__(32) void k_attn(const float* __restrict__ qkv,
                                             bf16* __restrict__ obf,
                                             int Nn, int E, int nh)
{
  int hd = E / nh;
  int n = blockIdx.x / nh, hh = blockIdx.x % nh;
  __shared__ float Q[32][32], Kk[32][32], V[32][32];
  int l = threadIdx.x;
  const float* base = qkv + (size_t)l * Nn * 3 * E + (size_t)n * 3 * E + hh * hd;
  for (int d = 0; d < hd; ++d) {
    Q[l][d]  = base[d];
    Kk[l][d] = base[E + d];
    V[l][d]  = base[2 * E + d];
  }
  __syncthreads();
  float scale = rsqrtf((float)hd);
  float sc[32];
  float mx = -1e30f;
  for (int t2 = 0; t2 < 32; ++t2) {
    float s = 0.f;
    for (int d = 0; d < hd; ++d) s += Q[l][d] * Kk[t2][d];
    s *= scale;
    sc[t2] = s;
    mx = fmaxf(mx, s);
  }
  float sum = 0.f;
  for (int t2 = 0; t2 < 32; ++t2) { sc[t2] = __expf(sc[t2] - mx); sum += sc[t2]; }
  float inv = 1.f / sum;
  for (int d = 0; d < hd; ++d) {
    float o = 0.f;
    for (int t2 = 0; t2 < 32; ++t2) o += sc[t2] * V[t2][d];
    obf[(size_t)l * Nn * E + (size_t)n * E + hh * hd + d] = (bf16)(o * inv);
  }
}

// ---------------------------------------------------------------------------
// LayerNorm over last dim (E in {128,256}); optional residual; wave per row.
// ---------------------------------------------------------------------------
__global__ __launch_bounds__(128) void k_ln(const float* __restrict__ x,
                                            const float* __restrict__ res,
                                            const float* __restrict__ g,
                                            const float* __restrict__ b,
                                            float* __restrict__ outF,
                                            bf16* __restrict__ outB, int E)
{
  int lane = threadIdx.x & 31, wave = threadIdx.x >> 5;
  size_t row = (size_t)blockIdx.x * 4 + wave;
  const float* xr = x + row * E;
  const float* rr = res ? res + row * E : nullptr;
  int per = E >> 5;
  float vals[8];
  float s = 0.f;
  for (int i = 0; i < per; ++i) {
    int idx = i * 32 + lane;
    float v = xr[idx] + (rr ? rr[idx] : 0.f);
    vals[i] = v; s += v;
  }
  for (int m = 16; m >= 1; m >>= 1) s += __shfl_xor(s, m, 32);
  float mean = s / (float)E;
  float vs = 0.f;
  for (int i = 0; i < per; ++i) { float d = vals[i] - mean; vs += d * d; }
  for (int m = 16; m >= 1; m >>= 1) vs += __shfl_xor(vs, m, 32);
  float inv = rsqrtf(vs / (float)E + 1e-5f);
  for (int i = 0; i < per; ++i) {
    int idx = i * 32 + lane;
    float o = (vals[i] - mean) * inv * g[idx] + b[idx];
    outF[row * E + idx] = o;
    if (outB) outB[row * E + idx] = (bf16)o;
  }
}

// ---------------------------------------------------------------------------
// Small utility kernels
// ---------------------------------------------------------------------------
__global__ void k_cvt(const float* __restrict__ in, bf16* __restrict__ out, size_t n) {
  size_t i = (size_t)blockIdx.x * 256 + threadIdx.x;
  if (i < n) out[i] = (bf16)in[i];
}

__global__ void k_fill(float* __restrict__ p, float v, size_t n) {
  size_t i = (size_t)blockIdx.x * 256 + threadIdx.x;
  if (i < n) p[i] = v;
}

// (Bn, R, Cc) -> (Bn, Cc, R), fp32 (+ optional bf16)
__global__ void k_transpose(const float* __restrict__ in, float* __restrict__ out,
                            bf16* __restrict__ outB, int Bn, int R, int Cc)
{
  size_t i = (size_t)blockIdx.x * 256 + threadIdx.x;
  size_t tot = (size_t)Bn * R * Cc;
  if (i >= tot) return;
  int c = (int)(i % Cc);
  size_t t = i / Cc;
  int r = (int)(t % R);
  int b = (int)(t / R);
  float v = in[i];
  size_t o = ((size_t)b * Cc + c) * R + r;
  out[o] = v;
  if (outB) outB[o] = (bf16)v;
}

__global__ void k_concat_bf(const float* __restrict__ a, const float* __restrict__ b,
                            bf16* __restrict__ o)
{
  int i = blockIdx.x * 256 + threadIdx.x;
  if (i >= 8192 * 256) return;
  int col = i & 255, row = i >> 8;
  float v = (col < 128) ? a[(size_t)row * 128 + col] : b[(size_t)row * 128 + col - 128];
  o[i] = (bf16)v;
}

// z0 = sum_t lin(X0, init_W, init_b); writes z and zT[:,0,:]
__global__ __launch_bounds__(128) void k_z0(const float* __restrict__ coeffs,
                                            const float* __restrict__ W,
                                            const float* __restrict__ b,
                                            float* __restrict__ z,
                                            float* __restrict__ zT)
{
  __shared__ float sx[64];
  int bi = blockIdx.x, tid = threadIdx.x;
  if (tid < 64) {
    int c = tid;
    const float* cb = coeffs + (size_t)bi * 255 * 256;
    float s = 0.f;
    for (int t = 0; t < 255; ++t) s += cb[(size_t)t * 256 + c];          // a rows
    for (int j = 0; j < 4; ++j) s += cb[(size_t)254 * 256 + j * 64 + c]; // extra last row
    sx[c] = s;
  }
  __syncthreads();
  if (tid < 128) {
    float s = 256.f * b[tid];
    const float* wr = W + tid * 64;
    for (int c = 0; c < 64; ++c) s += sx[c] * wr[c];
    z[bi * 128 + tid] = s;
    zT[(size_t)bi * 256 * 128 + tid] = s;
  }
}

// dx_dt (B,T,C) full array
__global__ void k_dxdt(const float* __restrict__ coeffs, float* __restrict__ dx)
{
  int i = blockIdx.x * 256 + threadIdx.x;
  if (i >= 32 * 256 * 64) return;
  int c = i & 63, tt = (i >> 6) & 255, b = i >> 14;
  const float* cb = coeffs + (size_t)b * 255 * 256;
  float v;
  if (tt < 255) v = cb[(size_t)tt * 256 + 64 + c];
  else {
    const float* l = cb + (size_t)254 * 256;
    v = l[64 + c] + 2.f * l[128 + c] + 3.f * l[192 + c];
  }
  dx[i] = v;
}

// RK4 combine: z += (k1+2k2+2k3+k4)/6 ; write zT slot t+1
__global__ void k_comb(float* __restrict__ z, const float* __restrict__ k1,
                       const float* __restrict__ k2, const float* __restrict__ k3,
                       const float* __restrict__ k4, float* __restrict__ zT, int t)
{
  int i = blockIdx.x * 128 + threadIdx.x;
  int b = i >> 7, h = i & 127;
  float zn = z[i] + (k1[i] + 2.f * k2[i] + 2.f * k3[i] + k4[i]) * (1.f / 6.f);
  z[i] = zn;
  zT[(size_t)b * 32768 + (size_t)(t + 1) * 128 + h] = zn;
}

// ---------------------------------------------------------------------------
// Host orchestration
// ---------------------------------------------------------------------------
extern "C" void kernel_launch(void* const* d_in, const int* in_sizes, int n_in,
                              void* d_out, int out_size, void* d_ws, size_t ws_size,
                              hipStream_t stream)
{
  (void)in_sizes; (void)n_in; (void)out_size; (void)ws_size;
  const float* x       = (const float*)d_in[0];
  const float* coeffs  = (const float*)d_in[1];
  const float* init_W  = (const float*)d_in[2];
  const float* init_b  = (const float*)d_in[3];
  const float* f0_W    = (const float*)d_in[4];
  const float* f0_b    = (const float*)d_in[5];
  const float* fmid_W  = (const float*)d_in[6];
  const float* fmid_b  = (const float*)d_in[7];
  const float* f4_W    = (const float*)d_in[8];
  const float* f4_b    = (const float*)d_in[9];
  const float* te_qkv_W = (const float*)d_in[10]; const float* te_qkv_b = (const float*)d_in[11];
  const float* te_out_W = (const float*)d_in[12]; const float* te_out_b = (const float*)d_in[13];
  const float* te_ln1_g = (const float*)d_in[14]; const float* te_ln1_b = (const float*)d_in[15];
  const float* te_ff1_W = (const float*)d_in[16]; const float* te_ff1_b = (const float*)d_in[17];
  const float* te_ff2_W = (const float*)d_in[18]; const float* te_ff2_b = (const float*)d_in[19];
  const float* te_ln2_g = (const float*)d_in[20]; const float* te_ln2_b = (const float*)d_in[21];
  const float* se_qkv_W = (const float*)d_in[22]; const float* se_qkv_b = (const float*)d_in[23];
  const float* se_out_W = (const float*)d_in[24]; const float* se_out_b = (const float*)d_in[25];
  const float* se_ln1_g = (const float*)d_in[26]; const float* se_ln1_b = (const float*)d_in[27];
  const float* se_ff1_W = (const float*)d_in[28]; const float* se_ff1_b = (const float*)d_in[29];
  const float* se_ff2_W = (const float*)d_in[30]; const float* se_ff2_b = (const float*)d_in[31];
  const float* se_ln2_g = (const float*)d_in[32]; const float* se_ln2_b = (const float*)d_in[33];
  const float* tr1_W = (const float*)d_in[34]; const float* tr1_b = (const float*)d_in[35];
  const float* tr2_W = (const float*)d_in[36]; const float* tr2_b = (const float*)d_in[37];
  const float* da1_W = (const float*)d_in[38]; const float* da1_b = (const float*)d_in[39];
  const float* da2_W = (const float*)d_in[40]; const float* da2_b = (const float*)d_in[41];
  const float* dr1_W = (const float*)d_in[42]; const float* dr1_b = (const float*)d_in[43];
  const float* dr2_W = (const float*)d_in[44]; const float* dr2_b = (const float*)d_in[45];

  float* out = (float*)d_out;
  const size_t OFF_RECON = 524288, OFF_SD = 1048576, OFF_RSD = 1572864;
  const size_t OFF_MASK = 2097152, OFF_ZT = 2105344;

  // bump allocator over d_ws
  char* wp = (char*)d_ws;
  auto alloc  = [&](size_t bytes) -> void* {
    void* p = (void*)wp; wp += (bytes + 255) & ~(size_t)255; return p;
  };
  auto allocF = [&](size_t n) { return (float*)alloc(n * 4); };
  auto allocB = [&](size_t n) { return (bf16*)alloc(n * 2); };
  auto cvt = [&](const float* s, bf16* d, size_t n) {
    k_cvt<<<dim3((unsigned)((n + 255) / 256)), dim3(256), 0, stream>>>(s, d, n);
  };
  auto gemm = [&](const bf16* A, const bf16* W, const float* bias,
                  float* oF, bf16* oB, int M, int N, int K, int act) {
    k_gemm<<<dim3(N / 64, M / 32), dim3(128), 0, stream>>>(A, W, bias, oF, oB, M, N, K, act);
  };
  auto ln = [&](const float* xx, const float* res, const float* g, const float* b,
                float* oF, bf16* oB, int R, int E) {
    k_ln<<<dim3(R / 4), dim3(128), 0, stream>>>(xx, res, g, b, oF, oB, E);
  };

  // -------- bf16 weights --------
  bf16* wf0 = allocB(256 * 128);   cvt(f0_W, wf0, 256 * 128);
  bf16* wm0 = allocB(256 * 256);   cvt(fmid_W, wm0, 256 * 256);
  bf16* wm1 = allocB(256 * 256);   cvt(fmid_W + 65536, wm1, 256 * 256);
  bf16* wf4 = allocB(8192 * 256);  cvt(f4_W, wf4, 8192 * 256);
  bf16* wqkv_te = allocB(768 * 256); cvt(te_qkv_W, wqkv_te, 768 * 256);
  bf16* wo_te   = allocB(256 * 256); cvt(te_out_W, wo_te, 256 * 256);
  bf16* wf1_te  = allocB(64 * 256);  cvt(te_ff1_W, wf1_te, 64 * 256);
  bf16* wf2_te  = allocB(256 * 64);  cvt(te_ff2_W, wf2_te, 256 * 64);
  bf16* wqkv_se = allocB(384 * 128); cvt(se_qkv_W, wqkv_se, 384 * 128);
  bf16* wo_se   = allocB(128 * 128); cvt(se_out_W, wo_se, 128 * 128);
  bf16* wf1_se  = allocB(64 * 128);  cvt(se_ff1_W, wf1_se, 64 * 128);
  bf16* wf2_se  = allocB(128 * 64);  cvt(se_ff2_W, wf2_se, 128 * 64);
  bf16* wtr1 = allocB(128 * 256); cvt(tr1_W, wtr1, 128 * 256);
  bf16* wtr2 = allocB(64 * 128);  cvt(tr2_W, wtr2, 64 * 128);
  bf16* wda1 = allocB(64 * 128);  cvt(da1_W, wda1, 64 * 128);
  bf16* wda2 = allocB(64 * 64);   cvt(da2_W, wda2, 64 * 64);
  bf16* wdr1 = allocB(128 * 64);  cvt(dr1_W, wdr1, 128 * 64);
  bf16* wdr2 = allocB(64 * 128);  cvt(dr2_W, wdr2, 64 * 128);

  // -------- state / scratch --------
  float* z    = allocF(4096);
  float* kk1  = allocF(4096);
  float* kk2  = allocF(4096);
  float* kk3  = allocF(4096);
  float* kk4  = allocF(4096);
  bf16*  h2s  = allocB(32 * 256);
  float* zT   = allocF(1048576);       // (B,256,H)
  bf16*  zTb  = allocB(1048576);
  float* dxdt = allocF(524288);        // (B,T,C)

  bf16*  hAb   = allocB(8192 * 256);
  bf16*  hBb   = allocB(8192 * 256);
  float* sd_pre = allocF(8192 * 128);
  bf16*  sdpb  = allocB(8192 * 128);
  bf16*  dab   = allocB(8192 * 64);
  bf16*  sdob  = allocB(8192 * 64);
  bf16*  drb   = allocB(8192 * 128);

  float* enc_qkv = allocF(3145728);    // max R*3E
  bf16*  enc_obf = allocB(1048576);
  float* enc_mha = allocF(1048576);
  float* enc_x1  = allocF(1048576);
  bf16*  enc_x1b = allocB(1048576);
  bf16*  enc_ff1b = allocB(524288);
  float* enc_ff2 = allocF(1048576);

  float* tAf = allocF(1048576); bf16* tAb = allocB(1048576);
  float* tBf = allocF(1048576); bf16* tBb = allocB(1048576);
  float* e1f = allocF(1048576); bf16* e1b = allocB(1048576);
  float* ftsf = allocF(1048576);
  float* esf  = allocF(1048576);
  float* fstf = allocF(1048576);
  bf16*  featsb = allocB(2097152);
  bf16*  trhb   = allocB(1048576);

  struct EncW {
    const bf16 *Wqkv, *Wo, *W1, *W2;
    const float *bqkv, *bo, *g1, *b1, *b1f, *b2f, *g2, *b2;
  };
  EncW TE = { wqkv_te, wo_te, wf1_te, wf2_te,
              te_qkv_b, te_out_b, te_ln1_g, te_ln1_b, te_ff1_b, te_ff2_b,
              te_ln2_g, te_ln2_b };
  EncW SE = { wqkv_se, wo_se, wf1_se, wf2_se,
              se_qkv_b, se_out_b, se_ln1_g, se_ln1_b, se_ff1_b, se_ff2_b,
              se_ln2_g, se_ln2_b };

  auto run_enc = [&](const float* xin, const bf16* xin_bf, int S, int Nn, int E, int nh,
                     const EncW& P, float* outF, bf16* outB) {
    int R = S * Nn;
    gemm(xin_bf, P.Wqkv, P.bqkv, enc_qkv, nullptr, R, 3 * E, E, 0);
    k_attn<<<dim3(Nn * nh), dim3(32), 0, stream>>>(enc_qkv, enc_obf, Nn, E, nh);
    gemm(enc_obf, P.Wo, P.bo, enc_mha, nullptr, R, E, E, 0);
    ln(enc_mha, xin, P.g1, P.b1, enc_x1, enc_x1b, R, E);
    gemm(enc_x1b, P.W1, P.b1f, nullptr, enc_ff1b, R, 64, E, 3);   // gelu
    gemm(enc_ff1b, P.W2, P.b2f, enc_ff2, nullptr, R, E, 64, 0);
    ln(enc_ff2, enc_x1, P.g2, P.b2, outF, outB, R, E);
  };

  // -------- z0, dx_dt --------
  k_z0<<<dim3(32), dim3(128), 0, stream>>>(coeffs, init_W, init_b, z, zT);
  k_dxdt<<<dim3((524288 + 255) / 256), dim3(256), 0, stream>>>(coeffs, dxdt);

  // -------- RK4 scan (255 steps, latency-bound, fused MLP + fused f4/contract) ----
  for (int t = 0; t < 255; ++t) {
    k_mlp3<<<1, 256, 0, stream>>>(z, nullptr, 0.f, wf0, f0_b, wm0, fmid_b, wm1, fmid_b + 256, h2s);
    k_f4c_scan<<<dim3(32), dim3(128), 0, stream>>>(h2s, wf4, f4_b, coeffs, t, 0.0f, kk1);
    k_mlp3<<<1, 256, 0, stream>>>(z, kk1, 0.5f, wf0, f0_b, wm0, fmid_b, wm1, fmid_b + 256, h2s);
    k_f4c_scan<<<dim3(32), dim3(128), 0, stream>>>(h2s, wf4, f4_b, coeffs, t, 0.5f, kk2);
    k_mlp3<<<1, 256, 0, stream>>>(z, kk2, 0.5f, wf0, f0_b, wm0, fmid_b, wm1, fmid_b + 256, h2s);
    k_f4c_scan<<<dim3(32), dim3(128), 0, stream>>>(h2s, wf4, f4_b, coeffs, t, 0.5f, kk3);
    k_mlp3<<<1, 256, 0, stream>>>(z, kk3, 1.0f, wf0, f0_b, wm0, fmid_b, wm1, fmid_b + 256, h2s);
    k_f4c_scan<<<dim3(32), dim3(128), 0, stream>>>(h2s, wf4, f4_b, coeffs, t, 1.0f, kk4);
    k_comb<<<dim3(32), dim3(128), 0, stream>>>(z, kk1, kk2, kk3, kk4, zT, t);
  }
  cvt(zT, zTb, 1048576);

  // -------- sd path: 8192 samples through _cde_f, fused f4 + contraction -----
  gemm(zTb, wf0, f0_b, nullptr, hAb, 8192, 256, 128, 1);
  gemm(hAb, wm0, fmid_b, nullptr, hBb, 8192, 256, 256, 1);
  gemm(hBb, wm1, fmid_b + 256, nullptr, hAb, 8192, 256, 256, 1);
  k_f4c_sd<<<dim3(256, 32), dim3(128), 0, stream>>>(hAb, wf4, f4_b, dxdt, sd_pre);
  cvt(sd_pre, sdpb, 8192 * 128);
  gemm(sdpb, wda1, da1_b, nullptr, dab, 8192, 64, 128, 1);
  gemm(dab, wda2, da2_b, out + OFF_SD, sdob, 8192, 64, 64, 0);
  gemm(sdob, wdr1, dr1_b, nullptr, drb, 8192, 128, 64, 1);
  gemm(drb, wdr2, dr2_b, out + OFF_RSD, nullptr, 8192, 64, 128, 0);

  // -------- encoders --------
  unsigned TG = (1048576 + 255) / 256;
  // features_time = enc_te( transpose(zT) : (32,128,256) )
  k_transpose<<<dim3(TG), dim3(256), 0, stream>>>(zT, tAf, tAb, 32, 256, 128);
  run_enc(tAf, tAb, 32, 128, 256, 8, TE, e1f, e1b);
  // features_time_spatial = enc_se( transpose(e1) : (32,256,128) )
  k_transpose<<<dim3(TG), dim3(256), 0, stream>>>(e1f, tBf, tBb, 32, 128, 256);
  run_enc(tBf, tBb, 32, 256, 128, 8, SE, ftsf, nullptr);
  // features_spatial = enc_se( zT : (32,256,128) )
  run_enc(zT, zTb, 32, 256, 128, 8, SE, esf, nullptr);
  // fst = transpose( enc_te( transpose(es) ) )
  k_transpose<<<dim3(TG), dim3(256), 0, stream>>>(esf, tAf, tAb, 32, 256, 128);
  run_enc(tAf, tAb, 32, 128, 256, 8, TE, e1f, nullptr);
  k_transpose<<<dim3(TG), dim3(256), 0, stream>>>(e1f, fstf, nullptr, 32, 128, 256);

  // -------- recon head --------
  k_concat_bf<<<dim3((2097152 + 255) / 256), dim3(256), 0, stream>>>(ftsf, fstf, featsb);
  gemm(featsb, wtr1, tr1_b, nullptr, trhb, 8192, 128, 256, 1);
  gemm(trhb, wtr2, tr2_b, out + OFF_RECON, nullptr, 8192, 64, 128, 0);

  // -------- passthrough outputs --------
  hipMemcpyAsync(out, x, 524288 * sizeof(float), hipMemcpyDeviceToDevice, stream);
  k_fill<<<dim3(8192 / 256), dim3(256), 0, stream>>>(out + OFF_MASK, 1.f, 8192);
  hipMemcpyAsync(out + OFF_ZT, zT, 1048576 * sizeof(float), hipMemcpyDeviceToDevice, stream);
}
